// NMTminGRUEncoder_30958124269568
// MI455X (gfx1250) — compile-verified, matrix operand order checked by reading, not run on hardware
//
#include <hip/hip_runtime.h>
#include <hip/hip_bf16.h>
#include <cstdint>
#include <cstddef>

// ---------------- problem constants (from reference) ----------------
#define B_   16
#define S_   2048
#define E_   512
#define H_   512
#define HI_  1024
#define V_   32000
#define M_   (B_ * S_)          // 32768 token rows

// ---------------- CDNA5 async global->LDS staging (feature-gated) ----------------
#if defined(__has_builtin)
# if __has_builtin(__builtin_amdgcn_global_load_async_to_lds_b128) && \
     __has_builtin(__builtin_amdgcn_s_wait_asynccnt)
#  define ASYNC_OK 1
# endif
#endif
#ifndef ASYNC_OK
# define ASYNC_OK 0
#endif

#define AS1 __attribute__((address_space(1)))
#define AS3 __attribute__((address_space(3)))

typedef int v4i_ls __attribute__((vector_size(16)));   // builtin's b128 payload type

// Copy 16 bytes global -> LDS. Async path issues GLOBAL_LOAD_ASYNC_TO_LDS_B128
// (tracked by ASYNCcnt, no VGPR round-trip); fallback is load+ds_store.
__device__ __forceinline__ void cp16_to_lds(const void* gsrc, void* ldst) {
#if ASYNC_OK
  AS1 void* g = (AS1 void*)gsrc;
  AS3 void* l = (AS3 void*)ldst;
  __builtin_amdgcn_global_load_async_to_lds_b128((AS1 v4i_ls*)g, (AS3 v4i_ls*)l,
                                                 /*offset=*/0, /*cpol=*/0);
#else
  *reinterpret_cast<uint4*>(ldst) = *reinterpret_cast<const uint4*>(gsrc);
#endif
}

// Drain this wave's ASYNCcnt, then block barrier (makes staged LDS visible).
__device__ __forceinline__ void stage_wait() {
#if ASYNC_OK
  __builtin_amdgcn_s_wait_asynccnt(0);
#endif
  __syncthreads();
}

// ---------------- WMMA plumbing (CDNA5 / gfx1250, wave32) ----------------
typedef __attribute__((ext_vector_type(16))) __bf16 v16bf;
typedef __attribute__((ext_vector_type(8)))  float  v8f;

union FragB16 { v16bf v; uint4 q[2]; };

__device__ __forceinline__ v8f wmma_bf16(v16bf a, v16bf b, v8f c) {
  // D = A(16x32 bf16) * B(32x16 bf16) + C(16x16 f32)
  return __builtin_amdgcn_wmma_f32_16x16x32_bf16(
      /*neg_a=*/false, a, /*neg_b=*/false, b,
      /*c_mod=*/(short)0, c, /*reuse_a=*/false, /*reuse_b=*/false);
}

// Row-major 16x32 bf16 tile -> WMMA operand fragment.
// Per ISA 7.12.2 (16-bit A 16x32): lane group g = lane>>4, row = lane&15;
//   frag elements 0..7  <- k = 8g .. 8g+7
//   frag elements 8..15 <- k = 16+8g .. 16+8g+7
// B operands use pre-transposed weights Wt[N][K] so the same loader applies
// (lane&15 selects the output column / Wt row).
__device__ __forceinline__ v16bf load_frag(const __hip_bfloat16* p, int ldm, int lane) {
  const int g = (lane >> 4) & 1;
  const int r = lane & 15;
  const __hip_bfloat16* row = p + (size_t)r * ldm;
  FragB16 f;
  f.q[0] = *reinterpret_cast<const uint4*>(row + 8 * g);
  f.q[1] = *reinterpret_cast<const uint4*>(row + 16 + 8 * g);
  return f.v;
}

__device__ __forceinline__ float softplus_f(float x) {
  // stable softplus with fast transcendentals (v_exp_f32 / v_log_f32)
  return fmaxf(x, 0.f) + __logf(1.f + __expf(-fabsf(x)));
}
__device__ __forceinline__ float log_g_f(float x) {
  return (x >= 0.f) ? __logf(x + 0.5f) : -softplus_f(-x);
}

// ---------------- one-time precision conversion / weight transpose ----------------
__global__ void k_cvt_bf16(const float* __restrict__ s, __hip_bfloat16* __restrict__ d, int n) {
  int i = blockIdx.x * blockDim.x + threadIdx.x;
  if (i < n) d[i] = __float2bfloat16(s[i]);
}

// src: [K,N] f32 row-major  ->  dst: [N,K] bf16 row-major (coalesced writes)
__global__ void k_cvt_t_bf16(const float* __restrict__ s, __hip_bfloat16* __restrict__ d,
                             int K, int N) {
  int i = blockIdx.x * blockDim.x + threadIdx.x;
  if (i < K * N) {
    int n = i / K, k = i % K;
    d[i] = __float2bfloat16(s[(size_t)k * N + n]);
  }
}

// ---------------- embedding gather + W_er GEMM (+bias), fused fwd & reversed ----------------
// One block per 16-row m-tile; 8 waves each own a 64-column slice of H=512.
__global__ __launch_bounds__(256) void k_embed_gemm(
    const int* __restrict__ xsrc, const int* __restrict__ xlen,
    const __hip_bfloat16* __restrict__ emb_bf,   // [V_, E_]
    const __hip_bfloat16* __restrict__ wer_t,    // [H_, E_]  (transposed)
    const float* __restrict__ b_er,
    __hip_bfloat16* __restrict__ xf,             // [M_, H_]
    __hip_bfloat16* __restrict__ xr) {           // [M_, H_] reversed+rotated stream
  __shared__ __align__(16) __hip_bfloat16 smA[16 * E_];   // 16 KB
  const int lane = threadIdx.x & 31;
  const int wv   = threadIdx.x >> 5;
  const int m_tile = blockIdx.x;                 // 0..2047
  // gather 16 embedding rows into LDS (async DMA, shared by all 8 waves)
  for (int i = threadIdx.x; i < 16 * E_ / 8; i += 256) {
    const int r  = i >> 6;                       // 64 x 16B chunks per 512-bf16 row
    const int cq = i & 63;
    const int tok = xsrc[m_tile * 16 + r];
    cp16_to_lds(emb_bf + (size_t)tok * E_ + cq * 8, smA + (size_t)i * 8);
  }
  stage_wait();

  const int n0 = wv * 64;
  const int g = lane >> 4, r15 = lane & 15;
  v8f acc[4] = {};
  const __hip_bfloat16* arow = smA + (size_t)r15 * E_;
  for (int k = 0; k < E_; k += 32) {
    FragB16 fa;
    fa.q[0] = *reinterpret_cast<const uint4*>(arow + k + 8 * g);
    fa.q[1] = *reinterpret_cast<const uint4*>(arow + k + 16 + 8 * g);
#pragma unroll
    for (int t = 0; t < 4; ++t) {
      v16bf fb = load_frag(wer_t + (size_t)(n0 + t * 16) * E_ + k, E_, lane);
      acc[t] = wmma_bf16(fa.v, fb, acc[t]);
    }
  }
  const int mbase = m_tile * 16;
  const int bidx  = mbase >> 11;                 // / S_  (tiles never straddle b)
  const int pad   = S_ - xlen[bidx];
#pragma unroll
  for (int t = 0; t < 4; ++t) {
    const int n = n0 + t * 16 + r15;
    const float bias = b_er[n];
#pragma unroll
    for (int e = 0; e < 8; ++e) {
      const int m = mbase + e + 8 * g;           // C layout: VGPR e -> row e + 8g
      const int s = m & (S_ - 1);
      // rev[b,t] = x_emb[b, S-1-((t+pad)%S)]  =>  t = (2S-1-s-pad) mod S
      const int trev = (2 * S_ - 1 - s - pad) & (S_ - 1);
      const __hip_bfloat16 bv = __float2bfloat16(acc[t][e] + bias);
      xf[(size_t)m * H_ + n] = bv;
      xr[((size_t)bidx * S_ + trev) * H_ + n] = bv;
    }
  }
}

// ---------------- x @ W_hg with fused log-space gate math ----------------
// Each wave computes a hidden tile (cols n0..) AND its paired gate tile
// (cols n0+HI..) sharing the A fragment, then writes lc/lv directly.
__global__ __launch_bounds__(256) void k_gemm_hg(
    const __hip_bfloat16* __restrict__ x,        // [M_, H_]
    const __hip_bfloat16* __restrict__ whg_t,    // [2*HI_, H_]: rows 0..HI-1 hidden, HI.. gate
    __hip_bfloat16* __restrict__ lc,             // [M_, HI_]
    __hip_bfloat16* __restrict__ lv) {           // [M_, HI_]
  __shared__ __align__(16) __hip_bfloat16 smA[16 * H_];   // 16 KB
  const int lane = threadIdx.x & 31;
  const int wv   = threadIdx.x >> 5;
  const int m_tile = blockIdx.x >> 2;            // 0..2047
  const int ngrp   = ((blockIdx.x & 3) << 3) + wv;  // 0..31 (32 col-pair groups)
  {
    const __hip_bfloat16* src = x + (size_t)m_tile * 16 * H_;
    for (int i = threadIdx.x; i < 16 * H_ / 8; i += 256)
      cp16_to_lds(src + (size_t)i * 8, smA + (size_t)i * 8);
  }
  stage_wait();

  const int n0 = ngrp * 32;
  const int g = lane >> 4, r15 = lane & 15;
  v8f acch[2] = {};
  v8f accg[2] = {};
  const __hip_bfloat16* arow = smA + (size_t)r15 * H_;
  for (int k = 0; k < H_; k += 32) {
    FragB16 fa;
    fa.q[0] = *reinterpret_cast<const uint4*>(arow + k + 8 * g);
    fa.q[1] = *reinterpret_cast<const uint4*>(arow + k + 16 + 8 * g);
#pragma unroll
    for (int t = 0; t < 2; ++t) {
      v16bf fbh = load_frag(whg_t + (size_t)(n0 + t * 16) * H_ + k, H_, lane);
      v16bf fbg = load_frag(whg_t + (size_t)(HI_ + n0 + t * 16) * H_ + k, H_, lane);
      acch[t] = wmma_bf16(fa.v, fbh, acch[t]);
      accg[t] = wmma_bf16(fa.v, fbg, accg[t]);
    }
  }
  const int mbase = m_tile * 16;
#pragma unroll
  for (int t = 0; t < 2; ++t) {
    const int n = n0 + t * 16 + r15;
#pragma unroll
    for (int e = 0; e < 8; ++e) {
      const int m = mbase + e + 8 * g;
      const float hh = acch[t][e];
      const float gg = accg[t][e];
      const size_t o = (size_t)m * HI_ + n;
      lc[o] = __float2bfloat16(-softplus_f(gg));                    // log(1-z)
      lv[o] = __float2bfloat16(-softplus_f(-gg) + log_g_f(hh));     // log z + log g(h~)
    }
  }
}

// ---------------- Heinsen scan: streaming cumsum + online log-sum-exp ----------------
// One thread per (batch, channel-pair); packed 32-bit bf16x2 traffic, fp32 state.
__global__ __launch_bounds__(256) void k_scan(
    const __hip_bfloat16* __restrict__ lc,
    const __hip_bfloat16* __restrict__ lv,
    __hip_bfloat16* __restrict__ hexp) {         // [M_, HI_] = exp(log_h)
  const int tid = blockIdx.x * blockDim.x + threadIdx.x;
  if (tid >= B_ * HI_ / 2) return;
  const int b  = tid >> 9;                       // / (HI_/2)
  const int c2 = (tid & (HI_ / 2 - 1)) * 2;
  float a0 = 0.f, M0 = -__builtin_inff(), R0 = 0.f;
  float a1 = 0.f, M1 = -__builtin_inff(), R1 = 0.f;
  size_t idx = (size_t)b * S_ * HI_ + c2;
  for (int s = 0; s < S_; ++s, idx += HI_) {
    if (s + 16 < S_) {                           // global_prefetch_b8 look-ahead
      __builtin_prefetch(lc + idx + (size_t)16 * HI_, 0, 0);
      __builtin_prefetch(lv + idx + (size_t)16 * HI_, 0, 0);
    }
    const __hip_bfloat162 pc = *reinterpret_cast<const __hip_bfloat162*>(lc + idx);
    const __hip_bfloat162 pv = *reinterpret_cast<const __hip_bfloat162*>(lv + idx);
    a0 += __bfloat162float(pc.x);
    a1 += __bfloat162float(pc.y);
    const float t0 = __bfloat162float(pv.x) - a0;
    const float t1 = __bfloat162float(pv.y) - a1;
    if (t0 > M0) { R0 = R0 * __expf(M0 - t0) + 1.f; M0 = t0; } else { R0 += __expf(t0 - M0); }
    if (t1 > M1) { R1 = R1 * __expf(M1 - t1) + 1.f; M1 = t1; } else { R1 += __expf(t1 - M1); }
    __hip_bfloat162 o;
    o.x = __float2bfloat16(__expf(a0 + M0 + __logf(R0)));
    o.y = __float2bfloat16(__expf(a1 + M1 + __logf(R1)));
    *reinterpret_cast<__hip_bfloat162*>(hexp + idx) = o;
  }
}

// ---------------- h_exp @ W_out; epilogue routes to next layer or d_out ----------------
// mode 0: bf16 -> next-layer x buffer
// mode 1: fp32 -> d_out[:, :, 0:H]        (final fwd)
// mode 2: fp32 -> d_out[:, un-rotate, H:] (final rev, flip/rotate fused into store)
__global__ __launch_bounds__(256) void k_gemm_out(
    const __hip_bfloat16* __restrict__ h,        // [M_, HI_]
    const __hip_bfloat16* __restrict__ wout_t,   // [H_, HI_] (transposed)
    __hip_bfloat16* __restrict__ xn,
    float* __restrict__ out,
    const int* __restrict__ xlen, int mode) {
  __shared__ __align__(16) __hip_bfloat16 smA[16 * HI_];  // 32 KB
  const int lane = threadIdx.x & 31;
  const int wv   = threadIdx.x >> 5;
  const int m_tile = blockIdx.x;                 // 0..2047
  {
    const __hip_bfloat16* src = h + (size_t)m_tile * 16 * HI_;
    for (int i = threadIdx.x; i < 16 * HI_ / 8; i += 256)
      cp16_to_lds(src + (size_t)i * 8, smA + (size_t)i * 8);
  }
  stage_wait();

  const int n0 = wv * 64;
  const int g = lane >> 4, r15 = lane & 15;
  v8f acc[4] = {};
  const __hip_bfloat16* arow = smA + (size_t)r15 * HI_;
  for (int k = 0; k < HI_; k += 32) {
    FragB16 fa;
    fa.q[0] = *reinterpret_cast<const uint4*>(arow + k + 8 * g);
    fa.q[1] = *reinterpret_cast<const uint4*>(arow + k + 16 + 8 * g);
#pragma unroll
    for (int t = 0; t < 4; ++t) {
      v16bf fb = load_frag(wout_t + (size_t)(n0 + t * 16) * HI_ + k, HI_, lane);
      acc[t] = wmma_bf16(fa.v, fb, acc[t]);
    }
  }
  const int mbase = m_tile * 16;
  const int bidx  = mbase >> 11;
  const int pad   = S_ - xlen[bidx];
#pragma unroll
  for (int t = 0; t < 4; ++t) {
    const int n = n0 + t * 16 + r15;
#pragma unroll
    for (int e = 0; e < 8; ++e) {
      const int m = mbase + e + 8 * g;
      const float v = acc[t][e];
      if (mode == 0) {
        xn[(size_t)m * H_ + n] = __float2bfloat16(v);
      } else if (mode == 1) {
        out[(size_t)m * (2 * H_) + n] = v;
      } else {
        const int s = m & (S_ - 1);
        const int trev = (2 * S_ - 1 - s - pad) & (S_ - 1);
        out[((size_t)bidx * S_ + trev) * (2 * H_) + H_ + n] = v;
      }
    }
  }
}

// ---------------- seq_h tail gather: out[b, len[b]-1, :] ----------------
__global__ void k_seqh(const int* __restrict__ xlen, float* __restrict__ out) {
  const int i = blockIdx.x * blockDim.x + threadIdx.x;
  if (i >= B_ * 2 * H_) return;
  const int b = i >> 10;                         // / (2H)
  const int j = i & (2 * H_ - 1);
  const size_t body = (size_t)M_ * 2 * H_;
  out[body + (size_t)b * 2 * H_ + j] =
      out[((size_t)b * S_ + (xlen[b] - 1)) * (2 * H_) + j];
}

// ---------------- host-side orchestration ----------------
extern "C" void kernel_launch(void* const* d_in, const int* in_sizes, int n_in,
                              void* d_out, int out_size, void* d_ws, size_t ws_size,
                              hipStream_t stream) {
  (void)in_sizes; (void)n_in; (void)out_size; (void)ws_size;
  const int*   xsrc   = (const int*)d_in[0];
  const int*   xlen   = (const int*)d_in[1];
  const float* emb    = (const float*)d_in[2];
  const float* wer    = (const float*)d_in[3];
  const float* ber    = (const float*)d_in[4];
  const float* whg_f  = (const float*)d_in[5];
  const float* wout_f = (const float*)d_in[6];
  const float* whg_r  = (const float*)d_in[7];
  const float* wout_r = (const float*)d_in[8];
  float* out = (float*)d_out;

  char* ws = (char*)d_ws;
  size_t off = 0;
  auto take = [&](size_t bytes) -> char* {
    char* p = ws + off;
    off = (off + bytes + 255) & ~(size_t)255;
    return p;
  };
  auto emb_bf = (__hip_bfloat16*)take((size_t)V_ * E_ * 2);
  auto wer_t  = (__hip_bfloat16*)take((size_t)H_ * E_ * 2);
  __hip_bfloat16* whg_t[2][3];
  __hip_bfloat16* wout_t[2][3];
  for (int d = 0; d < 2; ++d)
    for (int l = 0; l < 3; ++l) {
      whg_t[d][l]  = (__hip_bfloat16*)take((size_t)2 * HI_ * H_ * 2);
      wout_t[d][l] = (__hip_bfloat16*)take((size_t)H_ * HI_ * 2);
    }
  __hip_bfloat16* xbuf[2][2];
  for (int d = 0; d < 2; ++d)
    for (int p = 0; p < 2; ++p)
      xbuf[d][p] = (__hip_bfloat16*)take((size_t)M_ * H_ * 2);
  auto lc   = (__hip_bfloat16*)take((size_t)M_ * HI_ * 2);
  auto lv   = (__hip_bfloat16*)take((size_t)M_ * HI_ * 2);
  auto hexp = (__hip_bfloat16*)take((size_t)M_ * HI_ * 2);

  // 1) precision conversion (emb) + transpose+convert (weights), every call
  {
    int n = V_ * E_;
    k_cvt_bf16<<<(n + 255) / 256, 256, 0, stream>>>(emb, emb_bf, n);
    n = E_ * H_;
    k_cvt_t_bf16<<<(n + 255) / 256, 256, 0, stream>>>(wer, wer_t, E_, H_);
    for (int l = 0; l < 3; ++l) {
      n = H_ * 2 * HI_;
      k_cvt_t_bf16<<<(n + 255) / 256, 256, 0, stream>>>(
          whg_f + (size_t)l * H_ * 2 * HI_, whg_t[0][l], H_, 2 * HI_);
      k_cvt_t_bf16<<<(n + 255) / 256, 256, 0, stream>>>(
          whg_r + (size_t)l * H_ * 2 * HI_, whg_t[1][l], H_, 2 * HI_);
      n = HI_ * H_;
      k_cvt_t_bf16<<<(n + 255) / 256, 256, 0, stream>>>(
          wout_f + (size_t)l * HI_ * H_, wout_t[0][l], HI_, H_);
      k_cvt_t_bf16<<<(n + 255) / 256, 256, 0, stream>>>(
          wout_r + (size_t)l * HI_ * H_, wout_t[1][l], HI_, H_);
    }
  }

  // 2) embedding + input projection (writes fwd and reversed/rotated streams)
  k_embed_gemm<<<M_ / 16, 256, 0, stream>>>(xsrc, xlen, emb_bf, wer_t, ber,
                                            xbuf[0][0], xbuf[1][0]);

  // 3) three minGRU layers, fwd (d=0) then rev (d=1)
  for (int d = 0; d < 2; ++d) {
    int cur = 0;
    for (int l = 0; l < 3; ++l) {
      k_gemm_hg<<<(M_ / 16) * 4, 256, 0, stream>>>(xbuf[d][cur], whg_t[d][l], lc, lv);
      k_scan<<<(B_ * HI_ / 2) / 256, 256, 0, stream>>>(lc, lv, hexp);
      const int mode = (l == 2) ? (d == 0 ? 1 : 2) : 0;
      k_gemm_out<<<M_ / 16, 256, 0, stream>>>(hexp, wout_t[d][l],
                                              xbuf[d][cur ^ 1], out, xlen, mode);
      cur ^= 1;
    }
  }

  // 4) seq_h gather into the output tail
  k_seqh<<<(B_ * 2 * H_) / 256, 256, 0, stream>>>(xlen, out);
}
